// RNTN_773094113845
// MI455X (gfx1250) — matrix operand order repaired and checked by reading, not compiled, not run on hardware
//
#include <hip/hip_runtime.h>
#include <hip/hip_bf16.h>

#define HID 128
#define XDIM 256            // 2*HID
#define NK   128            // number of V slices (== HID)

typedef __attribute__((ext_vector_type(16))) __bf16 v16bf;
typedef __attribute__((ext_vector_type(8)))  float  v8f;

union Frag {
    v16bf v;
    uint4 q[2];
    unsigned short h[16];
};

// ---- bf16 helpers (round-to-nearest-even), no reliance on __bf16 scalar math ----
__device__ __forceinline__ unsigned short f2bf(float f) {
    unsigned u = __float_as_uint(f);
    unsigned r = u + 0x7FFFu + ((u >> 16) & 1u);
    return (unsigned short)(r >> 16);
}
__device__ __forceinline__ float bf2f(unsigned short h) {
    return __uint_as_float(((unsigned)h) << 16);
}

// ---------------------------------------------------------------------------
// 1) Convert V (128 x 256 x 256 f32, row-major) into bf16 blocks pre-swizzled
//    into the WMMA A-operand lane/slot layout (16-bit A 16x32, ISA 7.12.2).
//    Block (k, it, jt) covers V[k][it*16 .. +16][jt*32 .. +32]; 512 bf16 =
//    32 lanes x 16 slots, stored lane-major so each lane loads 32 contiguous B.
// ---------------------------------------------------------------------------
__global__ __launch_bounds__(256) void rntn_convert_v(
    const float* __restrict__ V, unsigned short* __restrict__ Vbf)
{
    int gid  = blockIdx.x * 256 + threadIdx.x;      // 0 .. 128*16*8*32-1
    int lane = gid & 31;
    int bk   = gid >> 5;                            // block index
    int jt   = bk & 7;
    int it   = (bk >> 3) & 15;
    int k    = bk >> 7;
    const float* src = V + (size_t)k * (XDIM * XDIM)
                         + (size_t)(it * 16 + (lane & 15)) * XDIM
                         + jt * 32;
    int khalf = (lane >> 4) * 8;                    // lanes 16-31 shift K by 8
    unsigned short out[16];
#pragma unroll
    for (int s = 0; s < 16; ++s) {
        int v  = s >> 1, hh = s & 1;
        int kk = ((v < 4) ? 0 : 16) + khalf + ((v & 3) * 2) + hh;
        out[s] = f2bf(src[kk]);
    }
    uint4* dst = (uint4*)Vbf + (size_t)gid * 2;     // 32 bytes per lane
    dst[0] = *(const uint4*)&out[0];
    dst[1] = *(const uint4*)&out[8];
}

// ---------------------------------------------------------------------------
// 2) Leaf init: H[t] = embed[word_ids[t]] for t in [0, n_leaves)
// ---------------------------------------------------------------------------
__global__ __launch_bounds__(HID) void rntn_leaf(
    const float* __restrict__ embed, const int* __restrict__ wid,
    float* __restrict__ H)
{
    int t = blockIdx.x;
    int j = threadIdx.x;
    H[(size_t)t * HID + j] = embed[(size_t)wid[t] * HID + j];
}

// ---------------------------------------------------------------------------
// 3) One tree level. grid = (128 k-slices, ceil(levelSize/16) node tiles).
//    block = 128 threads = 4 waves; wave w owns i-tiles {w, w+4, w+8, w+12}.
//    Computes H[t][k] = tanh( x^T V[k] x + (x W)[k] + b[k] ) fully fused.
// ---------------------------------------------------------------------------
__global__ __launch_bounds__(128) void rntn_level(
    float* __restrict__ H, const unsigned short* __restrict__ Vbf,
    const float* __restrict__ W, const float* __restrict__ b,
    const int* __restrict__ left, const int* __restrict__ right,
    int levelStart, int levelSize)
{
    __shared__ __align__(16) unsigned short XT[XDIM][16]; // XT[j][n], bf16
    __shared__ float sP[4][32];
    __shared__ int   lr[2][16];

    const int tid  = threadIdx.x;
    const int k    = blockIdx.x;
    const int base = levelStart + blockIdx.y * 16;

    if (tid < 16) {
        int t = base + tid;
        if (t >= levelStart + levelSize) t = levelStart;  // clamp padding
        lr[0][tid] = left[t];
        lr[1][tid] = right[t];
    }
    __syncthreads();

    // Build XT[j][n] = bf16(x[node n][j]) where x = concat(H[l], H[r]).
    {
        int nn = tid >> 3;            // node 0..15
        int g  = tid & 7;
        const float* hl = H + (size_t)lr[0][nn] * HID;
        const float* hr = H + (size_t)lr[1][nn] * HID;
#pragma unroll 4
        for (int i = 0; i < 16; ++i) {
            int j = g + 8 * i;        // 0..127
            XT[j]       [nn] = f2bf(hl[j]);
            XT[j + HID] [nn] = f2bf(hr[j]);
        }
    }
    __syncthreads();

    const int wave = tid >> 5, lane = tid & 31;
    const int nIdx = lane & 15, hi = lane >> 4;

    // Hoist all 8 B fragments (X^T, lane%16 = N = node, slots = K = j in table order) once.
    Frag B[8];
#pragma unroll
    for (int jt = 0; jt < 8; ++jt) {
#pragma unroll
        for (int s = 0; s < 16; ++s)
            B[jt].h[s] = XT[jt * 32 + ((s < 8) ? 0 : 16) + 8 * hi + (s & 7)][nIdx];
    }

    const uint4* Ablk = (const uint4*)(Vbf + (size_t)k * 16 * 8 * 512);
    float acc = 0.f;

    for (int itc = 0; itc < 4; ++itc) {
        const int it = wave + itc * 4;
        if (itc < 3)   // prefetch next A i-tile (global_prefetch_b8 path)
            __builtin_prefetch((const void*)(Ablk + (size_t)((it + 4) * 8) * 64 + lane * 2), 0, 1);

        v8f c = {};
#pragma unroll
        for (int jt = 0; jt < 8; ++jt) {
            Frag A;
            const uint4* pa = Ablk + (size_t)(it * 8 + jt) * 64 + lane * 2;
            A.q[0] = pa[0];
            A.q[1] = pa[1];
            c = __builtin_amdgcn_wmma_f32_16x16x32_bf16(
                    false, A.v, false, B[jt].v, (short)0, c, false, false);
        }
        // Second contraction: s[n] += sum_m x[n, i0+m] * U[i0+m, n]
        const int i0 = it * 16;
#pragma unroll
        for (int r = 0; r < 8; ++r) {
            int m = r + 8 * hi;                       // C/D layout: VGPR r -> row
            acc += c[r] * bf2f(XT[i0 + m][nIdx]);
        }
    }

    sP[wave][lane] = acc;
    __syncthreads();

    if (tid < 16) {
        int nn = tid;
        int t  = base + nn;
        if (t < levelStart + levelSize) {
            float s = b[k];
#pragma unroll
            for (int w = 0; w < 4; ++w) s += sP[w][nn] + sP[w][nn + 16];
            // linear term x @ W column k (W is (256,128) row-major) — broadcast reads
#pragma unroll 8
            for (int j = 0; j < XDIM; ++j) s += bf2f(XT[j][nn]) * W[j * HID + k];
            H[(size_t)t * HID + k] = tanhf(s);
        }
    }
}

// ---------------------------------------------------------------------------
// 4) Head: logits = H @ Wout^T + bout, then log_softmax over the 5 classes.
// ---------------------------------------------------------------------------
__global__ __launch_bounds__(64) void rntn_head(
    const float* __restrict__ H, const float* __restrict__ Wout,
    const float* __restrict__ bout, float* __restrict__ out, int n)
{
    int t = blockIdx.x * 64 + threadIdx.x;
    if (t >= n) return;
    const float* h = H + (size_t)t * HID;
    float lg[5];
    float mx = -1e30f;
#pragma unroll
    for (int o = 0; o < 5; ++o) {
        float s = bout[o];
        for (int q = 0; q < HID; ++q) s += h[q] * Wout[o * HID + q];
        lg[o] = s;
        mx = fmaxf(mx, s);
    }
    float se = 0.f;
#pragma unroll
    for (int o = 0; o < 5; ++o) se += expf(lg[o] - mx);
    float lse = mx + logf(se);
#pragma unroll
    for (int o = 0; o < 5; ++o) out[(size_t)t * 5 + o] = lg[o] - lse;
}

// ---------------------------------------------------------------------------
extern "C" void kernel_launch(void* const* d_in, const int* in_sizes, int n_in,
                              void* d_out, int out_size, void* d_ws, size_t ws_size,
                              hipStream_t stream) {
    const float* embed = (const float*)d_in[0];
    const float* V     = (const float*)d_in[1];
    const float* W     = (const float*)d_in[2];
    const float* b     = (const float*)d_in[3];
    const float* Wout  = (const float*)d_in[4];
    const float* bout  = (const float*)d_in[5];
    const int*   wid   = (const int*)d_in[6];
    const int*   left  = (const int*)d_in[7];
    const int*   right = (const int*)d_in[8];
    // is_leaf (d_in[9]) unused: leaves are nodes [0, n_leaves) by construction.

    const int n       = in_sizes[6];          // 1023
    const int nleaves = (n + 1) / 2;          // 512

    // workspace layout: [ V bf16 swizzled : 16.78 MB ][ H : n*128 f32 ]
    const size_t vbf_bytes = (size_t)NK * 16 * 8 * 512 * sizeof(unsigned short);
    unsigned short* Vbf = (unsigned short*)d_ws;
    float* H = (float*)((char*)d_ws + vbf_bytes);

    // 1) V -> bf16 WMMA-A layout (one pass; V stays L2-resident afterwards)
    rntn_convert_v<<<(NK * 16 * 8 * 32) / 256, 256, 0, stream>>>(V, Vbf);

    // 2) leaves
    rntn_leaf<<<nleaves, HID, 0, stream>>>(embed, wid, H);

    // 3) internal levels, bottom-up; stream order is the level barrier
    int start = nleaves;
    int sz    = nleaves / 2;
    while (sz >= 1) {
        dim3 grid(NK, (sz + 15) / 16);
        rntn_level<<<grid, 128, 0, stream>>>(H, Vbf, W, b, left, right, start, sz);
        start += sz;
        sz >>= 1;
    }

    // 4) logits + log_softmax
    rntn_head<<<(n + 63) / 64, 64, 0, stream>>>(H, Wout, bout, (float*)d_out, n);
}
